// ShiftedWindowAttention_53077205844394
// MI455X (gfx1250) — compile-verified, matrix operand order checked by reading, not run on hardware
//
#include <hip/hip_runtime.h>

// ---------------------------------------------------------------------------
// Shifted-window attention (Swin), fully fused per-window kernel for gfx1250.
// One workgroup (8 wave32) handles one 7x7 window (padded to 64 tokens).
// All GEMMs use v_wmma_f32_16x16x32_f16; operands staged in LDS.
// Weights are DMA'd into LDS with GLOBAL_LOAD_ASYNC_TO_LDS_B64 (ASYNCcnt),
// overlapped with VALU staging / WMMA work.
// ---------------------------------------------------------------------------

typedef __attribute__((ext_vector_type(16))) _Float16 v16h;
typedef __attribute__((ext_vector_type(8)))  float    v8f;

#define TILE_LD 130   // halves; 64x130 per X/Q/K/O tile (65 dwords -> conflict free)
#define VT_LD    66   // halves; Vt is [128 channels][64 tokens]
#define S_LD     52   // floats; scores [4][64][52] (only cols <49 used)
#define P_LD     66   // halves; probs  [4][64][66]
#define W_LD    132   // halves; staged weights (264B rows: 8B-aligned for async B64)

// LDS map (bytes):
//   sX  @      0 : 3 * 64*130 * 2 = 49920   (Xq/Xk/Xv -> later Q/K, then O)
//   sVt @  49920 : 128*66 * 2     = 16896
//   sS  @  66816 : 4*64*52 * 4    = 53248
//   sP  @ 120064 : 4*64*66 * 2    = 33792   -> total 153856
//   sW  overlay @ 49920 : 3*128*132*2 = 101376 (phase 1-2 only, ends 151296)
//   sWp overlay @ 66816 : 128*132*2   =  33792 (phase 5-6, overlays dead sS)
#define SMEM_BYTES 153856

union FragU { v16h v; unsigned int u[8]; };

// Load a 16x32 f16 WMMA operand fragment from an i-major matrix (stride ld
// halves). For A: i = M row.  For B: i = N column with source stored [n][k].
// Per CDNA5 ISA: lanes 0-15 hold K {0..7,16..23}, lanes 16-31 hold K {8..15,24..31},
// two K values packed per dword.
__device__ __forceinline__ v16h load_frag(const _Float16* base, int ld, int i0, int k0) {
  const int lane = threadIdx.x & 31;
  const _Float16* p = base + (i0 + (lane & 15)) * ld + k0 + ((lane >> 4) << 3);
  const unsigned int* a = (const unsigned int*)p;
  const unsigned int* b = (const unsigned int*)(p + 16);
  FragU f;
  f.u[0] = a[0]; f.u[1] = a[1]; f.u[2] = a[2]; f.u[3] = a[3];
  f.u[4] = b[0]; f.u[5] = b[1]; f.u[6] = b[2]; f.u[7] = b[3];
  return f.v;
}

__device__ __forceinline__ v8f wmma16(v16h a, v16h b, v8f c) {
  return __builtin_amdgcn_wmma_f32_16x16x32_f16(false, a, false, b, (short)0, c, false, false);
}

// Async DMA: global -> LDS, 8 bytes per lane, tracked by ASYNCcnt.
__device__ __forceinline__ void async_copy_b64(unsigned lds_off, const void* gptr) {
  asm volatile("global_load_async_to_lds_b64 %0, %1, off"
               :: "v"(lds_off), "v"(gptr) : "memory");
}
__device__ __forceinline__ void wait_async0() {
  asm volatile("s_wait_asynccnt 0x0" ::: "memory");
}

__device__ __forceinline__ int regid(int p) { return p < 49 ? 0 : (p < 53 ? 1 : 2); }

// Convert the 4 weight matrices (q,k,v,proj), each 128x128 f32 row-major
// ([c_out][c_in]) to f16 in workspace. Row index = N (c_out), col = K (c_in):
// exactly the [n][k] layout load_frag wants for the B operand.
__global__ void swin_cvt_weights(const float* __restrict__ qw, const float* __restrict__ kw,
                                 const float* __restrict__ vw, const float* __restrict__ pw,
                                 _Float16* __restrict__ w16) {
  int i = blockIdx.x * 256 + threadIdx.x;            // 0 .. 65535
  const float* s = (i < 16384) ? qw : (i < 32768) ? kw : (i < 49152) ? vw : pw;
  w16[i] = (_Float16)s[i & 16383];
}

__global__ __launch_bounds__(256) void swin_window_attn(
    const float* __restrict__ in_q, const float* __restrict__ in_k, const float* __restrict__ in_v,
    const _Float16* __restrict__ w16,
    const float* __restrict__ qb, const float* __restrict__ kb,
    const float* __restrict__ vb, const float* __restrict__ pb,
    const float* __restrict__ rpb, float* __restrict__ out) {
  __shared__ __align__(16) unsigned char smem[SMEM_BYTES];
  _Float16* sX  = (_Float16*)smem;                  // 3 tiles of 64 x TILE_LD
  _Float16* sVt = (_Float16*)(smem + 49920);
  float*    sS  = (float*)   (smem + 66816);
  _Float16* sP  = (_Float16*)(smem + 120064);
  _Float16* sW  = (_Float16*)(smem + 49920);        // overlay (phase 1-2 only)
  _Float16* sWp = (_Float16*)(smem + 66816);        // overlay on sS (phase 5-6)

  const int blk  = blockIdx.x;           // b*64 + wy*8 + wx
  const int b    = blk >> 6;
  const int wy   = (blk >> 3) & 7;
  const int wx   = blk & 7;
  const int tid  = threadIdx.x;
  const int wid  = tid >> 5;
  const int lane = tid & 31;
  const int lhalf = lane >> 4;
  const int l16   = lane & 15;

  v8f zero;
#pragma unroll
  for (int i = 0; i < 8; ++i) zero[i] = 0.f;

  // ---- Phase 1a: kick off async DMA of Wq/Wk/Wv (f16) into LDS -------------
  // 3*128 rows of 256B (dense in d_ws) -> 264B-stride rows in LDS.
  {
    const char* wbase = (const char*)w16;
    for (int e = tid; e < 3 * 128 * 32; e += 256) {       // 8B chunks
      int row = e >> 5, c = e & 31;
      unsigned dst = (unsigned)(size_t)(sW + row * W_LD) + c * 8;
      async_copy_b64(dst, wbase + e * 8);
    }
  }

  // ---- Phase 1b: stage input tiles (shift-roll gather, f32 -> f16) ---------
  for (int g = 0; g < 3; ++g) {
    const float* src = (g == 0) ? in_q : (g == 1) ? in_k : in_v;
    _Float16* dst = sX + g * (64 * TILE_LD);
    for (int e = tid; e < 64 * 128; e += 256) {
      int row = e >> 7, c = e & 127;
      float v = 0.f;
      if (row < 49) {
        int i = row / 7, j = row - (row / 7) * 7;
        int h = wy * 7 + i + 3; if (h >= 56) h -= 56;   // roll by -shift
        int w = wx * 7 + j + 3; if (w >= 56) w -= 56;
        v = __builtin_nontemporal_load(&src[(((size_t)b * 56 + h) * 56 + w) * 128 + c]);
      }
      dst[row * TILE_LD + c] = (_Float16)v;
    }
  }
  wait_async0();
  __syncthreads();

  // ---------------- Phase 2: Q/K/V projections (64x128 @ 128x128 each) ------
  const int mt  = wid >> 1;          // M tile 0..3
  const int nt0 = (wid & 1) * 4;     // N tiles nt0..nt0+3
  v8f aq[4], ak[4], av[4];
#pragma unroll
  for (int j = 0; j < 4; ++j) { aq[j] = zero; ak[j] = zero; av[j] = zero; }
#pragma unroll
  for (int kt = 0; kt < 4; ++kt) {
    const int k0 = kt * 32;
    v16h aQ = load_frag(sX,                 TILE_LD, mt * 16, k0);
    v16h aK = load_frag(sX +  64 * TILE_LD, TILE_LD, mt * 16, k0);
    v16h aV = load_frag(sX + 128 * TILE_LD, TILE_LD, mt * 16, k0);
#pragma unroll
    for (int j = 0; j < 4; ++j) {
      const int n0 = (nt0 + j) * 16;
      v16h bQ = load_frag(sW,              W_LD, n0, k0);
      v16h bK = load_frag(sW + 128 * W_LD, W_LD, n0, k0);
      v16h bV = load_frag(sW + 256 * W_LD, W_LD, n0, k0);
      aq[j] = wmma16(aQ, bQ, aq[j]);
      ak[j] = wmma16(aK, bK, ak[j]);
      av[j] = wmma16(aV, bV, av[j]);
    }
  }
  __syncthreads();   // all reads of sX/sW done before overwriting
  {
    const float qscale = 0.17677669529663687f;   // 32^-0.5
#pragma unroll
    for (int j = 0; j < 4; ++j) {
      const int n = (nt0 + j) * 16 + l16;
      const float bq_ = qb[n], bk_ = kb[n], bv_ = vb[n];
#pragma unroll
      for (int r = 0; r < 8; ++r) {
        const int m = mt * 16 + r + lhalf * 8;
        sX[m * TILE_LD + n]                = (_Float16)((aq[j][r] + bq_) * qscale);
        sX[64 * TILE_LD + m * TILE_LD + n] = (_Float16)(ak[j][r] + bk_);
        sVt[n * VT_LD + m]                 = (_Float16)(av[j][r] + bv_);   // V transposed
      }
    }
  }
  __syncthreads();

  // ---------------- Phase 3: scores = Qh @ Kh^T  (+rpb +shift mask) --------
  {
    const int head = wid >> 1;
    const int mtb  = (wid & 1) * 2;
    const _Float16* Q = sX;
    const _Float16* K = sX + 64 * TILE_LD;
#pragma unroll
    for (int mi = 0; mi < 2; ++mi) {
      const int m0 = (mtb + mi) * 16;
      v16h aF = load_frag(Q, TILE_LD, m0, head * 32);
#pragma unroll
      for (int nt = 0; nt < 4; ++nt) {
        v16h bF = load_frag(K, TILE_LD, nt * 16, head * 32);
        v8f s = wmma16(aF, bF, zero);
        const int n = nt * 16 + l16;
        if (n < 49) {
          const int ki = n / 7, kj = n - (n / 7) * 7;
          const int rk = regid(wy * 7 + ki) * 3 + regid(wx * 7 + kj);
#pragma unroll
          for (int r = 0; r < 8; ++r) {
            const int m = m0 + r + lhalf * 8;
            float val = s[r];
            if (m < 49) {
              const int qi = m / 7, qj = m - (m / 7) * 7;
              val += rpb[((qi - ki + 6) * 13 + (qj - kj + 6)) * 4 + head];
              const int rq = regid(wy * 7 + qi) * 3 + regid(wx * 7 + qj);
              if (rq != rk) val -= 100.f;
            }
            sS[head * 64 * S_LD + m * S_LD + n] = val;
          }
        }
      }
    }
  }
  __syncthreads();

  // ---------------- Phase 4: row softmax (1 thread per head-row) ------------
  {
    const int head2 = tid >> 6;
    const int row   = tid & 63;
    const float* srow = sS + head2 * 64 * S_LD + row * S_LD;
    _Float16* prow = sP + head2 * 64 * P_LD + row * P_LD;
    float mx = -3.0e38f;
    for (int j = 0; j < 49; ++j) mx = fmaxf(mx, srow[j]);
    float sum = 0.f;
    for (int j = 0; j < 49; ++j) {
      float e = __expf(srow[j] - mx);
      sum += e;
      prow[j] = (_Float16)e;                       // unnormalized
    }
    for (int j = 49; j < 64; ++j) prow[j] = (_Float16)0.f;
    const float inv = 1.0f / sum;
    for (int j = 0; j < 49; ++j)
      prow[j] = (_Float16)((float)prow[j] * inv);  // normalize in place
  }
  __syncthreads();

  // ---------------- Phase 5: async-stage Wproj; O_h = P @ V_h ---------------
  {
    const char* wbase = (const char*)(w16 + 3 * 128 * 128);
    for (int e = tid; e < 128 * 32; e += 256) {          // 8B chunks
      int row = e >> 5, c = e & 31;
      unsigned dst = (unsigned)(size_t)(sWp + row * W_LD) + c * 8;
      async_copy_b64(dst, wbase + e * 8);                // overlaps P@V below
    }
    const int head = wid >> 1;
    const int mtb  = (wid & 1) * 2;
    const _Float16* P  = sP  + head * 64 * P_LD;
    const _Float16* Vh = sVt + head * 32 * VT_LD;
#pragma unroll
    for (int mi = 0; mi < 2; ++mi) {
      const int m0 = (mtb + mi) * 16;
#pragma unroll
      for (int nt = 0; nt < 2; ++nt) {
        v8f acc = zero;
#pragma unroll
        for (int kt = 0; kt < 2; ++kt) {
          v16h aF = load_frag(P,  P_LD,  m0,      kt * 32);
          v16h bF = load_frag(Vh, VT_LD, nt * 16, kt * 32);
          acc = wmma16(aF, bF, acc);
        }
        const int n = head * 32 + nt * 16 + l16;
#pragma unroll
        for (int r = 0; r < 8; ++r) {
          const int m = m0 + r + lhalf * 8;
          sX[m * TILE_LD + n] = (_Float16)acc[r];   // O overwrites Q tile
        }
      }
    }
  }
  wait_async0();
  __syncthreads();

  // ---------------- Phase 6: out = O @ Wp^T + b ; inverse-roll scatter ------
  {
    const int mt6 = wid >> 1;
    const int nt6 = (wid & 1) * 4;
    v8f acc[4];
#pragma unroll
    for (int j = 0; j < 4; ++j) acc[j] = zero;
#pragma unroll
    for (int kt = 0; kt < 4; ++kt) {
      v16h aF = load_frag(sX, TILE_LD, mt6 * 16, kt * 32);
#pragma unroll
      for (int j = 0; j < 4; ++j) {
        v16h bF = load_frag(sWp, W_LD, (nt6 + j) * 16, kt * 32);
        acc[j] = wmma16(aF, bF, acc[j]);
      }
    }
#pragma unroll
    for (int j = 0; j < 4; ++j) {
      const int n = (nt6 + j) * 16 + l16;
      const float bias = pb[n];
#pragma unroll
      for (int r = 0; r < 8; ++r) {
        const int m = mt6 * 16 + r + lhalf * 8;
        if (m < 49) {
          const int i  = m / 7, jj = m - (m / 7) * 7;
          int h = wy * 7 + i + 3;  if (h >= 56) h -= 56;   // roll back by +shift
          int w = wx * 7 + jj + 3; if (w >= 56) w -= 56;
          __builtin_nontemporal_store(acc[j][r] + bias,
              &out[(((size_t)b * 56 + h) * 56 + w) * 128 + n]);
        }
      }
    }
  }
}

extern "C" void kernel_launch(void* const* d_in, const int* in_sizes, int n_in,
                              void* d_out, int out_size, void* d_ws, size_t ws_size,
                              hipStream_t stream) {
  const float* in_q = (const float*)d_in[0];
  const float* in_k = (const float*)d_in[1];
  const float* in_v = (const float*)d_in[2];
  const float* qw   = (const float*)d_in[3];
  const float* qb   = (const float*)d_in[4];
  const float* kw   = (const float*)d_in[5];
  const float* kb   = (const float*)d_in[6];
  const float* vw   = (const float*)d_in[7];
  const float* vb   = (const float*)d_in[8];
  const float* pw   = (const float*)d_in[9];
  const float* pb   = (const float*)d_in[10];
  const float* rpb  = (const float*)d_in[11];
  float* out = (float*)d_out;

  _Float16* w16 = (_Float16*)d_ws;   // 4 * 128*128 f16 = 128 KB

  swin_cvt_weights<<<256, 256, 0, stream>>>(qw, kw, vw, pw, w16);
  swin_window_attn<<<2048, 256, 0, stream>>>(in_q, in_k, in_v, w16,
                                             qb, kb, vb, pb, rpb, out);
}
